// CausalSelfAttention_1116691497247
// MI455X (gfx1250) — matrix operand order repaired and checked by reading, not compiled, not run on hardware
//
#include <hip/hip_runtime.h>
#include <hip/hip_bf16.h>

#define BB 8
#define TT 1024
#define CC 768
#define HH 12
#define DD 64
#define MM (BB * TT)      // 8192
#define N3 (3 * CC)       // 2304

typedef __attribute__((ext_vector_type(16))) __bf16 v16bf;
typedef __attribute__((ext_vector_type(8)))  float  v8f;
typedef __attribute__((ext_vector_type(4)))  unsigned int u32x4;
typedef __attribute__((ext_vector_type(4)))  int i32x4;
typedef __attribute__((ext_vector_type(8)))  int i32x8;

struct BF16Frag { union { v16bf v; u32x4 u[2]; }; };

// A-fragment (16x32, MxK): lane holds row m=lane%16; half=lane/16 selects
// K = half*16 + j, j=0..15 (contiguous).
__device__ __forceinline__ BF16Frag load_frag_a(const __bf16* rowptr, int half) {
    BF16Frag f;
    f.u[0] = *reinterpret_cast<const u32x4*>(rowptr + half * 16);
    f.u[1] = *reinterpret_cast<const u32x4*>(rowptr + half * 16 + 8);
    return f;
}

// B-fragment (32x16, KxN) from a TRANSPOSED operand BT[n][k]: lane holds
// col n=lane%16; k = half*16 + j, j=0..15 contiguous.
__device__ __forceinline__ BF16Frag load_frag_bT(const __bf16* rowptr, int half) {
    BF16Frag f;
    f.u[0] = *reinterpret_cast<const u32x4*>(rowptr + half * 16);
    f.u[1] = *reinterpret_cast<const u32x4*>(rowptr + half * 16 + 8);
    return f;
}

__device__ __forceinline__ v8f wmma_bf16(const BF16Frag& a, const BF16Frag& b, v8f c) {
    return __builtin_amdgcn_wmma_f32_16x16x32_bf16(
        /*neg_a=*/false, a.v, /*neg_b=*/false, b.v,
        /*c_mod=*/(short)0, c, /*reuse_a=*/false, /*reuse_b=*/false);
}

// The TDM builtin carries no pointer operand, so LLVM cannot see that it
// writes LDS; without this, reads of the staged buffer fold to undef.
// Escape the buffer into an asm with a memory clobber to model the DMA write.
__device__ __forceinline__ void tdm_lds_publish(const __bf16* p) {
    asm volatile("" :: "v"(p) : "memory");
}

// ---------------------------------------------------------------------------
// TDM: DMA a 64(rows) x 128(K) bf16 tile of a [rows x 768] row-major tensor
// into LDS at byte offset lds_off. TDM pad feature inserts 4 DWORDs (16 B)
// after every 64 DWORDs (= one 128-elem row) -> LDS row stride 136 elems,
// spreading banks with zero VALU cost.
// D# layout per CDNA5 ISA ch.8 (group0: count/lds/global/type;
// group1: data_size, pad ctl, dims, strides). Groups 2/3 unused (2D tensor).
// ---------------------------------------------------------------------------
#define KCHUNK   128
#define NCHUNK   (CC / KCHUNK)          // 6
#define BROWS    64
#define BLDS_LD  (KCHUNK + 8)           // 136 elements per LDS row
#define BUFELEMS (BROWS * BLDS_LD)      // 8704 elems = 17408 B per buffer

__device__ __forceinline__ void tdm_load_btile(unsigned lds_off_bytes,
                                               const __bf16* gtile,
                                               unsigned tensor_rows) {
    unsigned long long ga = (unsigned long long)gtile;
    u32x4 g0;
    g0[0] = 1u;                                   // count=1, user mode
    g0[1] = lds_off_bytes;                        // lds_addr
    g0[2] = (unsigned)(ga & 0xffffffffu);         // global_addr[31:0]
    g0[3] = (unsigned)(ga >> 32) | (2u << 30);    // global_addr[56:32] | type=2
    i32x8 g1;
    g1[0] = (int)((1u << 16)        // data_size = 1 (2 bytes)
                | (1u << 20)        // pad_enable
                | (5u << 22)        // pad_interval: every 64 DWORDs
                | (3u << 25));      // pad_amount: 4 DWORDs
    g1[1] = (int)((unsigned)CC << 16);            // tensor_dim0 = 768 (lo16 @63:48)
    g1[2] = (int)(tensor_rows << 16);             // tensor_dim1 (lo16 @95:80)
    g1[3] = (int)((unsigned)KCHUNK << 16);        // tile_dim0 = 128 (@127:112)
    g1[4] = BROWS;                                // tile_dim1 = 64 (@143:128)
    g1[5] = CC;                                   // tensor_dim0_stride = 768
    g1[6] = (int)((unsigned)CC << 16);            // tensor_dim1_stride lo16
    g1[7] = 0;
    i32x4 z4 = {0, 0, 0, 0};
    i32x8 z8 = {0, 0, 0, 0, 0, 0, 0, 0};
    __builtin_amdgcn_tensor_load_to_lds(g0, g1, z4, z4, z8, 0);
}

// ---------------------------------------------------------------------------
// Kernel 1: fp32 -> bf16 conversion; weights transposed to [N,K] for
// contiguous B-fragment / TDM-row loads.
// ---------------------------------------------------------------------------
__global__ void __launch_bounds__(256) convert_kernel(
    const float* __restrict__ x, const float* __restrict__ Wa,
    const float* __restrict__ Wp,
    __bf16* __restrict__ Xb, __bf16* __restrict__ WaT, __bf16* __restrict__ WpT) {
    const int NX  = MM * CC;
    const int NWA = CC * N3;
    const int NWP = CC * CC;
    int total = NX + NWA + NWP;
    for (int i = blockIdx.x * blockDim.x + threadIdx.x; i < total;
         i += gridDim.x * blockDim.x) {
        if (i < NX) {
            Xb[i] = (__bf16)x[i];
        } else if (i < NX + NWA) {
            int j = i - NX;
            int k = j / N3, n = j % N3;
            WaT[(size_t)n * CC + k] = (__bf16)Wa[j];
        } else {
            int j = i - NX - NWA;
            int k = j / CC, n = j % CC;
            WpT[(size_t)n * CC + k] = (__bf16)Wp[j];
        }
    }
}

// ---------------------------------------------------------------------------
// Kernel 2: QKV GEMM. Block = 128(M) x 64(N); 8 waves each own 16 M-rows and
// share a TDM-staged, double-buffered B panel in LDS. Epilogue adds bias and
// scatters to Q[B,H,T,D], K[B,H,T,D], V^T[B,H,D,T].
// ---------------------------------------------------------------------------
__global__ void __launch_bounds__(256) gemm_qkv_kernel(
    const __bf16* __restrict__ Xb, const __bf16* __restrict__ WaT,
    const float* __restrict__ b_attn,
    __bf16* __restrict__ Qb, __bf16* __restrict__ Kb, __bf16* __restrict__ VbT) {
    __shared__ __bf16 bsm[2 * BUFELEMS];

    int lane = threadIdx.x & 31, wave = threadIdx.x >> 5;
    const int tilesN = N3 / 64;               // 36
    int tbm = blockIdx.x / tilesN, tbn = blockIdx.x % tilesN;
    int m0 = tbm * 128 + wave * 16, n0 = tbn * 64;
    int half = lane >> 4, l15 = lane & 15;

    v8f acc[4] = {};
    const __bf16* arow  = Xb + (size_t)(m0 + l15) * CC;
    const __bf16* btile = WaT + (size_t)n0 * CC;

    if (wave == 0) tdm_load_btile(0, btile, (unsigned)(N3 - n0));

    for (int c = 0; c < NCHUNK; ++c) {
        if (c + 1 < NCHUNK) {
            if (wave == 0) {
                tdm_load_btile((unsigned)(((c + 1) & 1) * BUFELEMS * 2),
                               btile + (c + 1) * KCHUNK, (unsigned)(N3 - n0));
                __builtin_amdgcn_s_wait_tensorcnt(1);   // chunk c landed
            }
        } else if (wave == 0) {
            __builtin_amdgcn_s_wait_tensorcnt(0);
        }
        __syncthreads();
        tdm_lds_publish(bsm);   // model the DMA's LDS write for the compiler

        const __bf16* bbuf = &bsm[(c & 1) * BUFELEMS];
        int kbase = c * KCHUNK;
#pragma unroll
        for (int kk = 0; kk < KCHUNK; kk += 32) {
            __builtin_prefetch(arow + kbase + kk + 64, 0, 0);
            BF16Frag a = load_frag_a(arow + kbase + kk, half);
#pragma unroll
            for (int s = 0; s < 4; ++s) {
                const __bf16* bp = bbuf + (s * 16 + l15) * BLDS_LD + kk;
                BF16Frag b;
                b.u[0] = *reinterpret_cast<const u32x4*>(bp + half * 16);
                b.u[1] = *reinterpret_cast<const u32x4*>(bp + half * 16 + 8);
                acc[s] = wmma_bf16(a, b, acc[s]);
            }
        }
        __syncthreads();   // before buffer reuse
    }

#pragma unroll
    for (int s = 0; s < 4; ++s) {
        int n = n0 + s * 16 + l15;
        int which = n / CC;                   // 0=q 1=k 2=v
        int c = n - which * CC;
        int hh = c >> 6, d = c & 63;
        float bias = b_attn[n];
#pragma unroll
        for (int r = 0; r < 8; ++r) {
            int Mrow = m0 + r + half * 8;
            int bb = Mrow >> 10, t = Mrow & 1023;
            __bf16 val = (__bf16)(acc[s][r] + bias);
            size_t hb = (size_t)(bb * HH + hh);
            if (which == 0)      Qb[(hb * TT + t) * DD + d] = val;
            else if (which == 1) Kb[(hb * TT + t) * DD + d] = val;
            else                 VbT[(hb * DD + d) * TT + t] = val;
        }
    }
}

// ---------------------------------------------------------------------------
// Kernel 3: flash attention. One wave per (b,h,16-row q block). 32-key blocks:
// 4 WMMAs for S = Q K^T, f32 online softmax (row reductions via shfl within
// 16-lane halves), P relayout via LDS, 4 WMMAs for PV accumulation.
// ---------------------------------------------------------------------------
__global__ void __launch_bounds__(256) attn_kernel(
    const __bf16* __restrict__ Qb, const __bf16* __restrict__ Kb,
    const __bf16* __restrict__ VbT, __bf16* __restrict__ Yb) {
    __shared__ __bf16 plds[8][16 * 40];       // padded 16x32 P tile per wave

    int lane = threadIdx.x & 31, wave = threadIdx.x >> 5;
    int gw = blockIdx.x * 8 + wave;           // 6144 waves
    int qb = gw & 63;
    int bh = gw >> 6;                         // b*H + h
    int hh = bh % HH, bb = bh / HH;
    int half = lane >> 4, l15 = lane & 15;
    int t0 = qb * 16;

    const __bf16* Qh = Qb  + (size_t)bh * (TT * DD);
    const __bf16* Kh = Kb  + (size_t)bh * (TT * DD);
    const __bf16* Vh = VbT + (size_t)bh * (DD * TT);

    BF16Frag qa0 = load_frag_a(Qh + (size_t)(t0 + l15) * DD + 0,  half);
    BF16Frag qa1 = load_frag_a(Qh + (size_t)(t0 + l15) * DD + 32, half);

    v8f y0 = {}, y1 = {}, y2 = {}, y3 = {};
    float mrow[8], lrow[8];
#pragma unroll
    for (int r = 0; r < 8; ++r) { mrow[r] = -3.0e38f; lrow[r] = 0.f; }

    __bf16* pl = &plds[wave][0];
    const float scale = 0.125f;               // 1/sqrt(64)

    int nkb = (t0 + 47) >> 5;                 // key blocks touching keys <= t0+15
    for (int kbi = 0; kbi < nkb; ++kbi) {
        int key0 = kbi << 5;
        // S = Q K^T  (two 16-key groups x two 32-deep d halves)
        BF16Frag k00 = load_frag_bT(Kh + (size_t)(key0 + l15) * DD + 0,  half);
        BF16Frag k01 = load_frag_bT(Kh + (size_t)(key0 + l15) * DD + 32, half);
        BF16Frag k10 = load_frag_bT(Kh + (size_t)(key0 + 16 + l15) * DD + 0,  half);
        BF16Frag k11 = load_frag_bT(Kh + (size_t)(key0 + 16 + l15) * DD + 32, half);
        v8f s0 = {}, s1 = {};
        s0 = wmma_bf16(qa0, k00, s0); s0 = wmma_bf16(qa1, k01, s0);
        s1 = wmma_bf16(qa0, k10, s1); s1 = wmma_bf16(qa1, k11, s1);

        int col0 = key0 + l15, col1 = col0 + 16;
#pragma unroll
        for (int r = 0; r < 8; ++r) {
            int trow = t0 + r + half * 8;
            float a0 = s0[r] * scale; if (col0 > trow) a0 = -3.0e38f;
            float a1 = s1[r] * scale; if (col1 > trow) a1 = -3.0e38f;
            // row max across the 16 lanes holding this row
            float mx = fmaxf(a0, a1);
            mx = fmaxf(mx, __shfl_xor(mx, 1, 16));
            mx = fmaxf(mx, __shfl_xor(mx, 2, 16));
            mx = fmaxf(mx, __shfl_xor(mx, 4, 16));
            mx = fmaxf(mx, __shfl_xor(mx, 8, 16));
            float mnew  = fmaxf(mrow[r], mx);
            float alpha = __expf(mrow[r] - mnew);
            float p0 = __expf(a0 - mnew);
            float p1 = __expf(a1 - mnew);
            float rs = p0 + p1;
            rs += __shfl_xor(rs, 1, 16);
            rs += __shfl_xor(rs, 2, 16);
            rs += __shfl_xor(rs, 4, 16);
            rs += __shfl_xor(rs, 8, 16);
            lrow[r] = lrow[r] * alpha + rs;
            mrow[r] = mnew;
            y0[r] *= alpha; y1[r] *= alpha; y2[r] *= alpha; y3[r] *= alpha;
            int prow = r + half * 8;
            pl[prow * 40 + l15]      = (__bf16)p0;
            pl[prow * 40 + 16 + l15] = (__bf16)p1;
        }
        // C-layout -> A-layout relayout through LDS; per-wave DS is in-order.
        asm volatile("s_wait_dscnt 0" ::: "memory");
        BF16Frag pf = load_frag_a(pl + l15 * 40, half);
        BF16Frag v0 = load_frag_bT(Vh + (size_t)(0 * 16 + l15) * TT + key0, half);
        BF16Frag v1 = load_frag_bT(Vh + (size_t)(1 * 16 + l15) * TT + key0, half);
        BF16Frag v2 = load_frag_bT(Vh + (size_t)(2 * 16 + l15) * TT + key0, half);
        BF16Frag v3 = load_frag_bT(Vh + (size_t)(3 * 16 + l15) * TT + key0, half);
        y0 = wmma_bf16(pf, v0, y0);
        y1 = wmma_bf16(pf, v1, y1);
        y2 = wmma_bf16(pf, v2, y2);
        y3 = wmma_bf16(pf, v3, y3);
    }

#pragma unroll
    for (int r = 0; r < 8; ++r) {
        int trow = t0 + r + half * 8;
        float inv = 1.0f / lrow[r];
        size_t base = ((size_t)(bb * TT + trow)) * CC + hh * 64 + l15;
        Yb[base + 0]  = (__bf16)(y0[r] * inv);
        Yb[base + 16] = (__bf16)(y1[r] * inv);
        Yb[base + 32] = (__bf16)(y2[r] * inv);
        Yb[base + 48] = (__bf16)(y3[r] * inv);
    }
}

// ---------------------------------------------------------------------------
// Kernel 4: output projection GEMM, same TDM-staged structure, f32 out + bias.
// ---------------------------------------------------------------------------
__global__ void __launch_bounds__(256) gemm_proj_kernel(
    const __bf16* __restrict__ Yb, const __bf16* __restrict__ WpT,
    const float* __restrict__ b_proj, float* __restrict__ out) {
    __shared__ __bf16 bsm[2 * BUFELEMS];

    int lane = threadIdx.x & 31, wave = threadIdx.x >> 5;
    const int tilesN = CC / 64;               // 12
    int tbm = blockIdx.x / tilesN, tbn = blockIdx.x % tilesN;
    int m0 = tbm * 128 + wave * 16, n0 = tbn * 64;
    int half = lane >> 4, l15 = lane & 15;

    v8f acc[4] = {};
    const __bf16* arow  = Yb + (size_t)(m0 + l15) * CC;
    const __bf16* btile = WpT + (size_t)n0 * CC;

    if (wave == 0) tdm_load_btile(0, btile, (unsigned)(CC - n0));

    for (int c = 0; c < NCHUNK; ++c) {
        if (c + 1 < NCHUNK) {
            if (wave == 0) {
                tdm_load_btile((unsigned)(((c + 1) & 1) * BUFELEMS * 2),
                               btile + (c + 1) * KCHUNK, (unsigned)(CC - n0));
                __builtin_amdgcn_s_wait_tensorcnt(1);
            }
        } else if (wave == 0) {
            __builtin_amdgcn_s_wait_tensorcnt(0);
        }
        __syncthreads();
        tdm_lds_publish(bsm);   // model the DMA's LDS write for the compiler

        const __bf16* bbuf = &bsm[(c & 1) * BUFELEMS];
        int kbase = c * KCHUNK;
#pragma unroll
        for (int kk = 0; kk < KCHUNK; kk += 32) {
            __builtin_prefetch(arow + kbase + kk + 64, 0, 0);
            BF16Frag a = load_frag_a(arow + kbase + kk, half);
#pragma unroll
            for (int s = 0; s < 4; ++s) {
                const __bf16* bp = bbuf + (s * 16 + l15) * BLDS_LD + kk;
                BF16Frag b;
                b.u[0] = *reinterpret_cast<const u32x4*>(bp + half * 16);
                b.u[1] = *reinterpret_cast<const u32x4*>(bp + half * 16 + 8);
                acc[s] = wmma_bf16(a, b, acc[s]);
            }
        }
        __syncthreads();
    }

#pragma unroll
    for (int s = 0; s < 4; ++s) {
        int n = n0 + s * 16 + l15;
        float bias = b_proj[n];
#pragma unroll
        for (int r = 0; r < 8; ++r) {
            int Mrow = m0 + r + half * 8;
            out[(size_t)Mrow * CC + n] = acc[s][r] + bias;
        }
    }
}

// ---------------------------------------------------------------------------
extern "C" void kernel_launch(void* const* d_in, const int* in_sizes, int n_in,
                              void* d_out, int out_size, void* d_ws, size_t ws_size,
                              hipStream_t stream) {
    const float* x      = (const float*)d_in[0];
    const float* W_attn = (const float*)d_in[1];
    const float* b_attn = (const float*)d_in[2];
    const float* W_proj = (const float*)d_in[3];
    const float* b_proj = (const float*)d_in[4];
    float* out = (float*)d_out;

    char* ws = (char*)d_ws;
    size_t off = 0;
    auto carve = [&](size_t elems) {
        __bf16* p = (__bf16*)(ws + off);
        off += ((elems * sizeof(__bf16)) + 255) & ~(size_t)255;
        return p;
    };
    __bf16* Xb  = carve((size_t)MM * CC);          // x in bf16
    __bf16* WaT = carve((size_t)N3 * CC);          // W_attn^T bf16
    __bf16* WpT = carve((size_t)CC * CC);          // W_proj^T bf16
    __bf16* Qb  = carve((size_t)BB * HH * TT * DD);
    __bf16* Kb  = carve((size_t)BB * HH * TT * DD);
    __bf16* VbT = carve((size_t)BB * HH * DD * TT);
    __bf16* Yb  = carve((size_t)MM * CC);          // attention output [B*T, C]
    (void)ws_size; (void)in_sizes; (void)n_in; (void)out_size;

    convert_kernel <<<1024, 256, 0, stream>>>(x, W_attn, W_proj, Xb, WaT, WpT);
    gemm_qkv_kernel<<<(MM / 128) * (N3 / 64), 256, 0, stream>>>(
        Xb, WaT, b_attn, Qb, Kb, VbT);
    attn_kernel    <<<BB * HH * (TT / 16) / 8, 256, 0, stream>>>(Qb, Kb, VbT, Yb);
    gemm_proj_kernel<<<(MM / 128) * (CC / 64), 256, 0, stream>>>(
        Yb, WpT, b_proj, out);
}
